// RealCPCEncoder_74466142978483
// MI455X (gfx1250) — matrix operand order repaired and checked by reading, not compiled, not run on hardware
//
#include <hip/hip_runtime.h>
#include <hip/hip_bf16.h>

typedef __attribute__((ext_vector_type(16))) __bf16 bf16x16;
typedef __attribute__((ext_vector_type(8)))  float  floatx8;
typedef unsigned short u16;
typedef unsigned int   u32;
typedef int v4i __attribute__((vector_size(16)));

#if __has_builtin(__builtin_amdgcn_global_load_async_to_lds_b128)
#define CPC_HAS_ASYNC_LDS 1
typedef __attribute__((address_space(1))) v4i cpc_gv4i;
typedef __attribute__((address_space(3))) v4i cpc_lv4i;
#endif

// ---------------------------------------------------------------- helpers ---

__device__ inline u32 bf16pair(float a, float b) {
  u32 ua = __float_as_uint(a), ub = __float_as_uint(b);
  ua = (ua + 0x7FFFu + ((ua >> 16) & 1u)) >> 16;   // RNE
  ub = (ub + 0x7FFFu + ((ub >> 16) & 1u)) >> 16;
  return (ub << 16) | (ua & 0xFFFFu);
}
__device__ inline u16 f2bf(float a) {
  u32 ua = __float_as_uint(a);
  return (u16)((ua + 0x7FFFu + ((ua >> 16) & 1u)) >> 16);
}
__device__ inline float fast_tanh(float x) {
#if __has_builtin(__builtin_amdgcn_tanhf)
  return __builtin_amdgcn_tanhf(x);
#else
  float e = __expf(2.f * x);
  return (e - 1.f) * __builtin_amdgcn_rcpf(e + 1.f);
#endif
}
__device__ inline float fast_sigmoid(float x) {
  return __builtin_amdgcn_rcpf(1.f + __expf(-x));
}
__device__ inline float gelu_tanh(float x) {
  float x3 = x * x * x;
  return 0.5f * x * (1.f + fast_tanh(0.7978845608028654f * (x + 0.044715f * x3)));
}
__device__ inline void cpc_wait_async0() {
#if __has_builtin(__builtin_amdgcn_s_wait_asynccnt)
  __builtin_amdgcn_s_wait_asynccnt(0);
#else
  asm volatile("s_wait_asynccnt 0x0" ::: "memory");
#endif
}

// A is 16x32 bf16 (per-lane K-striped), B is 32x16 bf16 (per-VGPR K-pair, lane=N)
__device__ inline floatx8 wmma_bf16x32(const u32* au, const u32* bu, floatx8 c) {
  union Ua { u32 u[8]; bf16x16 v; } A, B;
#pragma unroll
  for (int i = 0; i < 8; ++i) { A.u[i] = au[i]; B.u[i] = bu[i]; }
  return __builtin_amdgcn_wmma_f32_16x16x32_bf16(false, A.v, false, B.v,
                                                 (short)0, c, false, false);
}
// q offset within a 32-wide K-tile for A-fragment VGPR v (contiguous K order, matches packed B)
__device__ inline int a_q(int v, int half) {
  return half * 16 + (v << 1);
}

// ---------------------------------------------------------------- sizes -----

#define BB   16
#define TT   32768
#define TD   2048
#define HH   512
#define MG   (TD * BB)          // 32768 GRU/proj rows (t-major: row = t*16+b)

constexpr size_t SZ_PRE = 16777216ull * 4;   // every layer has M*C = 16.7M elems
constexpr size_t SZ_H   = 16777216ull * 2;
constexpr size_t OFF_PRE = 0;
constexpr size_t OFF_H0  = OFF_PRE + SZ_PRE;
constexpr size_t OFF_H1  = OFF_H0 + SZ_H;
constexpr size_t OFF_H2  = OFF_H1 + SZ_H;
constexpr size_t OFF_H3T = OFF_H2 + SZ_H;
constexpr size_t OFF_G   = OFF_H3T + SZ_H;
constexpr size_t OFF_GI  = OFF_G + SZ_H;
constexpr size_t SZ_GI   = (size_t)MG * 1536 * 4;
constexpr size_t OFF_Z   = OFF_GI + SZ_GI;
constexpr size_t SZ_Z    = (size_t)MG * 512 * 4;
constexpr size_t OFF_WB1 = OFF_Z + SZ_Z;                 // conv1: NT=8  KT=18
constexpr size_t OFF_WB2 = OFF_WB1 + 8ull  * 18 * 256 * 4;
constexpr size_t OFF_WB3 = OFF_WB2 + 16ull * 36 * 256 * 4;
constexpr size_t OFF_WBI = OFF_WB3 + 32ull * 72 * 256 * 4;
constexpr size_t OFF_WBP = OFF_WBI + 96ull * 16 * 256 * 4;
constexpr size_t OFF_CF  = OFF_WBP + 32ull * 16 * 256 * 4;
constexpr size_t OFF_CB  = OFF_CF + 2ull * 16 * 512 * 4;
constexpr size_t OFF_PS  = OFF_CB + 2ull * 16 * 512 * 2;
constexpr size_t OFF_CTR = OFF_PS + 2048;

// ---------------------------------------------------------------- kernels ---

__global__ void cpc_init_kernel(float* cf, u16* cb, u32* ctr) {
  int i = blockIdx.x * 256 + threadIdx.x;
  if (i < 2 * 16 * 512) { cf[i] = 0.f; cb[i] = 0; }
  if (i == 0) *ctr = 0;
}

// Pack fp32 weight [K][N] into WMMA B-fragment stream: wB[nt][kt][v][lane]
__global__ void cpc_pack_b_kernel(const float* __restrict__ w,
                                  const float* __restrict__ cs,
                                  u32* __restrict__ wB, int KT, int N, int total) {
  int idx = blockIdx.x * 256 + threadIdx.x;
  if (idx >= total) return;
  int lane = idx & 31, v = (idx >> 5) & 7;
  int kt = (idx >> 8) % KT, nt = (idx >> 8) / KT;
  int K = kt * 32 + ((lane >> 4) << 4) + 2 * v;
  int col = nt * 16 + (lane & 15);
  float s = cs ? cs[col] : 1.f;
  wB[idx] = bf16pair(w[(size_t)K * N + col] * s, w[(size_t)(K + 1) * N + col] * s);
}

__global__ void cpc_proj_scale_kernel(const float* __restrict__ v,
                                      const float* __restrict__ g,
                                      float* __restrict__ s) {
  int c = threadIdx.x;                 // block of 512
  float acc = 0.f;
  for (int r = 0; r < 512; ++r) { float x = v[(size_t)r * 512 + c]; acc += x * x; }
  s[c] = g[c] / (sqrtf(acc) + 1e-8f);
}

// conv0: Cin=1 -> scalar fp32 (tiny), SAME stride2 pad_lo=3
__global__ void cpc_conv0_kernel(const float* __restrict__ x,
                                 const float* __restrict__ w,
                                 float* __restrict__ pre) {
  int idx = blockIdx.x * 256 + threadIdx.x;   // (b, t, c) over 16*16384*64
  int c = idx & 63, t = (idx >> 6) & 16383, b = idx >> 20;
  float s = 0.f;
#pragma unroll
  for (int k = 0; k < 9; ++k) {
    int ti = 2 * t - 3 + k;
    if (ti >= 0 && ti < TT) s += x[(size_t)b * TT + ti] * 1e20f * w[k * 64 + c];
  }
  pre[idx] = s;
}

// rmsnorm(x+bias)*scale -> gelu -> bf16; optional [b][t]->[t*16+b] transpose
template <int C, bool TRANS>
__global__ void cpc_rms_gelu_kernel(const float* __restrict__ pre,
                                    const float* __restrict__ bias,
                                    const float* __restrict__ scale,
                                    u16* __restrict__ out, int Td) {
  __shared__ float red[C];
  int row = blockIdx.x, c = threadIdx.x;
  float v = pre[(size_t)row * C + c] + bias[c];
  red[c] = v * v;
  __syncthreads();
  for (int off = C / 2; off > 0; off >>= 1) {
    if (c < off) red[c] += red[c + off];
    __syncthreads();
  }
  float rms = rsqrtf(red[0] / (float)C + 1e-6f);
  float y = gelu_tanh(v * rms * scale[c]);
  size_t drow = row;
  if (TRANS) { int b = row / Td, t = row % Td; drow = (size_t)t * 16 + b; }
  out[drow * C + c] = f2bf(y);
}

// Implicit-GEMM conv (stride 2, K=9, SAME pad_lo=3) with WMMA bf16.
// A = patches [M][9*CIN] staged via LDS, B = packed weight stream.
// Interior blocks stage via GLOBAL_LOAD_ASYNC_TO_LDS_B128 (ASYNCcnt path).
template <int CIN, int COUT, int TIN, int TOUT>
__global__ __launch_bounds__(256) void cpc_conv_wmma_kernel(
    const u16* __restrict__ in, const u32* __restrict__ wB, float* __restrict__ pre) {
  constexpr int MB = 32, ROWS = 2 * MB + 8, K9 = 9 * CIN, KT = K9 / 32;
  constexpr int NT = COUT / 16, NPW = NT / 4;
  __shared__ u16 in_s[ROWS * CIN];
  int b = blockIdx.y, t0 = blockIdx.x * MB;
  int gr0 = 2 * t0 - 3;
  bool interior = (gr0 >= 0) && (gr0 + ROWS <= TIN);
#ifdef CPC_HAS_ASYNC_LDS
  if (interior) {
    // contiguous ROWS*CIN bf16 span: DMA global -> LDS, 16B per op
    const u16* gp0 = in + ((size_t)b * TIN + gr0) * CIN;
    for (int ci = threadIdx.x; ci < ROWS * CIN / 8; ci += 256) {
      int i = ci * 8;
      __builtin_amdgcn_global_load_async_to_lds_b128(
          (cpc_gv4i*)(gp0 + i), (cpc_lv4i*)(&in_s[i]), 0, 0);
    }
    cpc_wait_async0();
  } else
#endif
  {
    for (int i2 = threadIdx.x; i2 < ROWS * CIN / 2; i2 += 256) {
      int i = i2 * 2;
      int lr = i / CIN, cin = i - lr * CIN;
      int gr = gr0 + lr;
      u32 val = 0;
      if (gr >= 0 && gr < TIN)
        val = *(const u32*)&in[((size_t)b * TIN + gr) * CIN + cin];
      *(u32*)&in_s[i] = val;
    }
  }
  __syncthreads();
  int wave = threadIdx.x >> 5, lane = threadIdx.x & 31;
  int mt = wave >> 2, ncol = wave & 3;
  int m = lane & 15, half = lane >> 4;
  for (int i = 0; i < NPW; ++i) {
    int nt = ncol * NPW + i;
    floatx8 acc = {0.f, 0.f, 0.f, 0.f, 0.f, 0.f, 0.f, 0.f};
    for (int kt = 0; kt < KT; ++kt) {
      u32 au[8], bu[8];
#pragma unroll
      for (int v = 0; v < 8; ++v) {
        int q = kt * 32 + a_q(v, half);
        int k = q / CIN, cin = q - k * CIN;
        int lr = 2 * (mt * 16 + m) + k;
        au[v] = *(const u32*)&in_s[lr * CIN + cin];
        bu[v] = wB[(((size_t)nt * KT + kt) * 8 + v) * 32 + lane];
      }
      acc = wmma_bf16x32(au, bu, acc);
    }
#pragma unroll
    for (int r = 0; r < 8; ++r) {
      int trow = t0 + mt * 16 + r + 8 * half;
      pre[((size_t)b * TOUT + trow) * COUT + nt * 16 + (lane & 15)] = acc[r];
    }
  }
}

// Generic bf16 WMMA GEMM: out[M][N] = A[M][K] * Wpacked + bias
template <int K, int N>
__global__ __launch_bounds__(256) void cpc_gemm_bf16_kernel(
    const u16* __restrict__ A, const u32* __restrict__ wB,
    const float* __restrict__ bias, float* __restrict__ out) {
  constexpr int KT = K / 32;
  int wave = threadIdx.x >> 5, lane = threadIdx.x & 31;
  int mt = wave >> 1, ntl = wave & 1;
  int Mbase = blockIdx.x * 64 + mt * 16;
  int ntglob = blockIdx.y * 2 + ntl;
  int m = lane & 15, half = lane >> 4;
  const u16* arow = A + (size_t)(Mbase + m) * K;
  floatx8 acc = {0.f, 0.f, 0.f, 0.f, 0.f, 0.f, 0.f, 0.f};
  for (int kt = 0; kt < KT; ++kt) {
    if (kt + 1 < KT)
      __builtin_prefetch(&wB[(((size_t)ntglob * KT + kt + 1) * 8) * 32 + lane], 0, 0);
    u32 au[8], bu[8];
#pragma unroll
    for (int v = 0; v < 8; ++v) {
      au[v] = *(const u32*)&arow[kt * 32 + a_q(v, half)];
      bu[v] = wB[(((size_t)ntglob * KT + kt) * 8 + v) * 32 + lane];
    }
    acc = wmma_bf16x32(au, bu, acc);
  }
  int col = ntglob * 16 + (lane & 15);
  float bv = bias ? bias[col] : 0.f;
#pragma unroll
  for (int r = 0; r < 8; ++r)
    out[(size_t)(Mbase + r + 8 * half) * N + col] = acc[r] + bv;
}

// Persistent GRU: 32 WGs, WG j owns H-slice [16j,16j+16). Wh slice pre-packed
// in LDS; c state double-buffered in global (bf16 for WMMA A, f32 for update);
// grid-wide step sync via L2 atomic counter (release fence before signal,
// acquire fence after wait so next step's c loads miss WGP$).
__global__ __launch_bounds__(128) void cpc_gru_kernel(
    const float* __restrict__ wh, const float* __restrict__ bhn,
    const float* __restrict__ gi, float* __restrict__ cf, u16* __restrict__ cb,
    u16* __restrict__ g, u32* __restrict__ ctr) {
  constexpr int NW = 32;
  __shared__ u32 whB[3 * 16 * 8 * 32];   // 48KB: [gate][kt][v][lane]
  __shared__ float gh_s[3 * 256];
  __shared__ float gi_s[3 * 256];
  __shared__ float bhn_s[16];
  int j = blockIdx.x, tid = threadIdx.x;
  for (int idx = tid; idx < 3 * 16 * 8 * 32; idx += 128) {
    int lane = idx & 31, v = (idx >> 5) & 7, kt = (idx >> 8) & 15, gate = idx >> 12;
    int K = kt * 32 + ((lane >> 4) << 4) + 2 * v;
    int col = gate * 512 + j * 16 + (lane & 15);
    whB[idx] = bf16pair(wh[(size_t)K * 1536 + col], wh[(size_t)(K + 1) * 1536 + col]);
  }
  if (tid < 16) bhn_s[tid] = bhn[j * 16 + tid];
  __syncthreads();
  int wave = tid >> 5, lane = tid & 31, m = lane & 15, half = lane >> 4;
  for (int t = 0; t < TD; ++t) {
    int rb = t & 1, wb = rb ^ 1;
    if (wave < 3) {
      const u16* crow = cb + (size_t)rb * 16 * 512 + (size_t)m * 512;
      floatx8 acc = {0.f, 0.f, 0.f, 0.f, 0.f, 0.f, 0.f, 0.f};
      for (int kt = 0; kt < 16; ++kt) {
        u32 au[8], bu[8];
#pragma unroll
        for (int v = 0; v < 8; ++v) {
          au[v] = *(const u32*)&crow[kt * 32 + a_q(v, half)];
          bu[v] = whB[((wave * 16 + kt) * 8 + v) * 32 + lane];
        }
        acc = wmma_bf16x32(au, bu, acc);
      }
#pragma unroll
      for (int r = 0; r < 8; ++r)
        gh_s[wave * 256 + (r + 8 * half) * 16 + (lane & 15)] = acc[r];
    } else {
      const float* gt = gi + (size_t)t * 16 * 1536;
      for (int idx = lane; idx < 768; idx += 32) {
        int gate = idx / 256, rem = idx - gate * 256, mm = rem >> 4, nn = rem & 15;
        gi_s[idx] = gt[(size_t)mm * 1536 + gate * 512 + j * 16 + nn];
      }
    }
    __syncthreads();
    for (int e = tid; e < 256; e += 128) {
      int mm = e >> 4, nn = e & 15, h = j * 16 + nn;
      float rg = fast_sigmoid(gi_s[e] + gh_s[e]);
      float zg = fast_sigmoid(gi_s[256 + e] + gh_s[256 + e]);
      float ng = fast_tanh(gi_s[512 + e] + rg * (gh_s[512 + e] + bhn_s[nn]));
      float cp = cf[(size_t)rb * 16 * 512 + mm * 512 + h];
      float cn = (1.f - zg) * ng + zg * cp;
      cf[(size_t)wb * 16 * 512 + mm * 512 + h] = cn;
      u16 cbv = f2bf(cn);
      cb[(size_t)wb * 16 * 512 + mm * 512 + h] = cbv;
      g[((size_t)t * 16 + mm) * 512 + h] = cbv;
    }
    __threadfence();            // release: make c/g stores visible device-wide
    __syncthreads();
    if (tid == 0) {
      atomicAdd(ctr, 1u);
      u32 target = (u32)(t + 1) * NW;
      while (*(volatile u32*)ctr < target) __builtin_amdgcn_s_sleep(2);
    }
    __syncthreads();
    __threadfence();            // acquire: invalidate near caches before reload
  }
}

// L2-normalize each z row and transpose [t*16+b] -> out[b][t][d]
__global__ __launch_bounds__(256) void cpc_norm_out_kernel(const float* __restrict__ z,
                                                           float* __restrict__ out) {
  __shared__ float red[256];
  int row = blockIdx.x, t = row >> 4, b = row & 15;
  const float* zr = z + (size_t)row * 512;
  float s = 0.f;
  for (int c = threadIdx.x; c < 512; c += 256) { float v = zr[c]; s += v * v; }
  red[threadIdx.x] = s;
  __syncthreads();
  for (int off = 128; off > 0; off >>= 1) {
    if (threadIdx.x < off) red[threadIdx.x] += red[threadIdx.x + off];
    __syncthreads();
  }
  float zn = sqrtf(red[0]);
  float inv = (zn > 1e-6f) ? 1.f / (zn + 1e-8f) : 1.f;
  float* orow = out + ((size_t)b * TD + t) * 512;
  for (int c = threadIdx.x; c < 512; c += 256) orow[c] = zr[c] * inv;
}

// ---------------------------------------------------------------- launch ----

extern "C" void kernel_launch(void* const* d_in, const int* in_sizes, int n_in,
                              void* d_out, int out_size, void* d_ws, size_t ws_size,
                              hipStream_t stream) {
  const float* x    = (const float*)d_in[0];
  const float* c0w  = (const float*)d_in[1];
  const float* c0b  = (const float*)d_in[2];
  const float* r0s  = (const float*)d_in[3];
  const float* c1w  = (const float*)d_in[4];
  const float* c1b  = (const float*)d_in[5];
  const float* r1s  = (const float*)d_in[6];
  const float* c2w  = (const float*)d_in[7];
  const float* c2b  = (const float*)d_in[8];
  const float* r2s  = (const float*)d_in[9];
  const float* c3w  = (const float*)d_in[10];
  const float* c3b  = (const float*)d_in[11];
  const float* r3s  = (const float*)d_in[12];
  const float* gwi  = (const float*)d_in[13];
  const float* gbi  = (const float*)d_in[14];
  const float* gwh  = (const float*)d_in[15];
  const float* gbhn = (const float*)d_in[16];
  const float* pv   = (const float*)d_in[17];
  const float* pg   = (const float*)d_in[18];
  const float* pb   = (const float*)d_in[19];

  char* ws = (char*)d_ws;
  float* pre  = (float*)(ws + OFF_PRE);
  u16*   h0   = (u16*)(ws + OFF_H0);
  u16*   h1   = (u16*)(ws + OFF_H1);
  u16*   h2   = (u16*)(ws + OFF_H2);
  u16*   h3t  = (u16*)(ws + OFF_H3T);
  u16*   gbf  = (u16*)(ws + OFF_G);
  float* gif  = (float*)(ws + OFF_GI);
  float* zf   = (float*)(ws + OFF_Z);
  u32*   wb1  = (u32*)(ws + OFF_WB1);
  u32*   wb2  = (u32*)(ws + OFF_WB2);
  u32*   wb3  = (u32*)(ws + OFF_WB3);
  u32*   wbi  = (u32*)(ws + OFF_WBI);
  u32*   wbp  = (u32*)(ws + OFF_WBP);
  float* cf   = (float*)(ws + OFF_CF);
  u16*   cb   = (u16*)(ws + OFF_CB);
  float* ps   = (float*)(ws + OFF_PS);
  u32*   ctr  = (u32*)(ws + OFF_CTR);

  cpc_init_kernel<<<64, 256, 0, stream>>>(cf, cb, ctr);

  // weight packing (bf16, WMMA-B layout)
  cpc_pack_b_kernel<<<(8 * 18 * 256 + 255) / 256, 256, 0, stream>>>(c1w, nullptr, wb1, 18, 128, 8 * 18 * 256);
  cpc_pack_b_kernel<<<(16 * 36 * 256 + 255) / 256, 256, 0, stream>>>(c2w, nullptr, wb2, 36, 256, 16 * 36 * 256);
  cpc_pack_b_kernel<<<(32 * 72 * 256 + 255) / 256, 256, 0, stream>>>(c3w, nullptr, wb3, 72, 512, 32 * 72 * 256);
  cpc_pack_b_kernel<<<(96 * 16 * 256 + 255) / 256, 256, 0, stream>>>(gwi, nullptr, wbi, 16, 1536, 96 * 16 * 256);
  cpc_proj_scale_kernel<<<1, 512, 0, stream>>>(pv, pg, ps);
  cpc_pack_b_kernel<<<(32 * 16 * 256 + 255) / 256, 256, 0, stream>>>(pv, ps, wbp, 16, 512, 32 * 16 * 256);

  // conv stack
  cpc_conv0_kernel<<<(16 * 16384 * 64) / 256, 256, 0, stream>>>(x, c0w, pre);
  cpc_rms_gelu_kernel<64, false><<<16 * 16384, 64, 0, stream>>>(pre, c0b, r0s, h0, 0);
  cpc_conv_wmma_kernel<64, 128, 16384, 8192><<<dim3(256, 16), 256, 0, stream>>>(h0, wb1, pre);
  cpc_rms_gelu_kernel<128, false><<<16 * 8192, 128, 0, stream>>>(pre, c1b, r1s, h1, 0);
  cpc_conv_wmma_kernel<128, 256, 8192, 4096><<<dim3(128, 16), 256, 0, stream>>>(h1, wb2, pre);
  cpc_rms_gelu_kernel<256, false><<<16 * 4096, 256, 0, stream>>>(pre, c2b, r2s, h2, 0);
  cpc_conv_wmma_kernel<256, 512, 4096, 2048><<<dim3(64, 16), 256, 0, stream>>>(h2, wb3, pre);
  cpc_rms_gelu_kernel<512, true><<<16 * 2048, 512, 0, stream>>>(pre, c3b, r3s, h3t, 2048);

  // gi = h3t @ Wi + bi  (t-major rows)
  cpc_gemm_bf16_kernel<512, 1536><<<dim3(MG / 64, 1536 / 32), 256, 0, stream>>>(h3t, wbi, gbi, gif);

  // recurrent GRU (persistent, 32 WGs)
  cpc_gru_kernel<<<32, 128, 0, stream>>>(gwh, gbhn, gif, cf, cb, gbf, ctr);

  // projection + output normalize
  cpc_gemm_bf16_kernel<512, 512><<<dim3(MG / 64, 512 / 32), 256, 0, stream>>>(gbf, wbp, pb, zf);
  cpc_norm_out_kernel<<<MG, 256, 0, stream>>>(zf, (float*)d_out);
}
